// SimpleBBoxHead_25228637897251
// MI455X (gfx1250) — matrix operand order repaired and checked by reading, not compiled, not run on hardware
//
#include <hip/hip_runtime.h>

// ---------------------------------------------------------------------------
// Types for CDNA5 WMMA (wave32, V_WMMA_F32_16X16X32_BF16)
// ---------------------------------------------------------------------------
typedef __attribute__((ext_vector_type(8)))  float   v8f;
typedef __attribute__((ext_vector_type(8)))  __bf16  v8bf;
typedef __attribute__((ext_vector_type(16))) __bf16  v16bf;
typedef int v4i_vs __attribute__((vector_size(16)));   // payload type of async b128 builtin

union V16BF {
    v16bf v;
    v8bf  h[2];
};

#define LDS_AS __attribute__((address_space(3)))
#define GLB_AS __attribute__((address_space(1)))

// f32 -> bf16 round-to-nearest-even (bit manip; no reliance on cast support)
static __device__ inline __bf16 f2bf(float f) {
    union { float f; unsigned u; } in;
    in.f = f;
    unsigned r = in.u + 0x7FFFu + ((in.u >> 16) & 1u);
    union { unsigned short s; __bf16 b; } out;
    out.s = (unsigned short)(r >> 16);
    return out.b;
}

static __device__ inline float bf2f(__bf16 b) {
    union { unsigned short s; __bf16 b; } in;
    in.b = b;
    union { float f; unsigned u; } out;
    out.u = ((unsigned)in.s) << 16;
    return out.f;
}

// ---------------------------------------------------------------------------
// CDNA5 async global->LDS copy (ASYNCcnt-tracked, no VGPR data movement).
// INST_OFFSET is added to BOTH the global and the LDS address (ISA 10.2),
// so one address pair serves a 32-byte stake via offset:0 / offset:16.
// Builtin signature (from hipcc diagnostic): (v4i AS1*, v4i AS3*, Ii, Ii).
// ---------------------------------------------------------------------------
template <int OFF>
static __device__ inline void async_b128(const __bf16* g, __bf16* l) {
#if __has_builtin(__builtin_amdgcn_global_load_async_to_lds_b128)
    __builtin_amdgcn_global_load_async_to_lds_b128(
        (GLB_AS v4i_vs*)g, (LDS_AS v4i_vs*)l, OFF, 0);
#else
    unsigned loff = (unsigned)(unsigned long long)(LDS_AS void*)l;
    unsigned long long ga = (unsigned long long)g;
    if constexpr (OFF == 0)
        asm volatile("global_load_async_to_lds_b128 %0, %1, off"
                     :: "v"(loff), "v"(ga) : "memory");
    else
        asm volatile("global_load_async_to_lds_b128 %0, %1, off offset:16"
                     :: "v"(loff), "v"(ga) : "memory");
#endif
}

static __device__ inline void wait_async0() {
#if __has_builtin(__builtin_amdgcn_s_wait_asynccnt)
    __builtin_amdgcn_s_wait_asynccnt(0);
#else
    asm volatile("s_wait_asynccnt 0x0" ::: "memory");
#endif
}

// ---------------------------------------------------------------------------
// Problem constants
// ---------------------------------------------------------------------------
#define LM_DIM 4096
#define V_DIM  1024
#define N_V    32
#define VIS_FLAT (V_DIM * N_V)          // 32768
#define IN_DIM   (VIS_FLAT + LM_DIM)    // 36864
#define FF     1024
#define BATCH  16
#define TOK    2048
#define OUTD   6

// ---------------------------------------------------------------------------
// Kernel 1: elementwise f32 -> bf16
// ---------------------------------------------------------------------------
__global__ __launch_bounds__(256)
void f32_to_bf16_kernel(const float* __restrict__ src, __bf16* __restrict__ dst, int n) {
    int i = blockIdx.x * 256 + threadIdx.x;
    if (i < n) dst[i] = f2bf(src[i]);
}

// ---------------------------------------------------------------------------
// Kernel 2: convert + transpose weights.
//   src: f32 [K][N] row-major (fan_in x fan_out, as in the reference)
//   dst: bf16 [N][K] row-major (so the GEMM's B tile is n-major, k contiguous)
// ---------------------------------------------------------------------------
__global__ __launch_bounds__(256)
void conv_transpose_kernel(const float* __restrict__ src, __bf16* __restrict__ dst,
                           int K, int N) {
    __shared__ float tile[32][33];
    const int k0 = blockIdx.x * 32;
    const int n0 = blockIdx.y * 32;
    const int c  = threadIdx.x & 31;
    const int r8 = threadIdx.x >> 5;   // 0..7
    #pragma unroll
    for (int i = 0; i < 4; ++i) {
        int r = r8 + i * 8;
        tile[r][c] = src[(size_t)(k0 + r) * N + (n0 + c)];
    }
    __syncthreads();
    #pragma unroll
    for (int i = 0; i < 4; ++i) {
        int r = r8 + i * 8;            // r indexes n in the output
        dst[(size_t)(n0 + r) * K + (k0 + c)] = f2bf(tile[c][r]);
    }
}

// ---------------------------------------------------------------------------
// Kernel 3: bf16 WMMA GEMM + bias + ReLU, out in bf16.
//   C[m][n] = relu( sum_k A[m][k] * W[k][n] + bias[n] )
//   Wt is bf16 [FF][K] (pre-transposed).
//   GATHER=true: A row t is concat(vis[bidx[t]][:], lm[t][:]) (layer 1).
//
// Tile 128x128, K-step 32, 256 threads = 8 waves (4M x 2N), each wave 2x4
// subtiles -> 8 v_wmma per step. Double-buffered LDS filled by
// GLOBAL_LOAD_ASYNC_TO_LDS_B128: one barrier + one s_wait_asynccnt per step,
// async DMA of tile s+1 overlaps WMMA on tile s.
// ---------------------------------------------------------------------------
template <bool GATHER>
__global__ __launch_bounds__(256)
void gemm_relu_kernel(const __bf16* __restrict__ A,
                      const __bf16* __restrict__ vis,   // [BATCH][VIS_FLAT]
                      const __bf16* __restrict__ lm,    // [TOK][LM_DIM]
                      const int*    __restrict__ bidx,  // [TOK]
                      const __bf16* __restrict__ Wt,    // [FF][K]
                      const float*  __restrict__ bias,  // [FF]
                      __bf16* __restrict__ out,         // [TOK][FF]
                      int K) {
    // [buffer][A=0/B=1][128 rows x 32 k]  -> 2 x 16 KB
    __shared__ __bf16 lds[2][2][128 * 32];

    const int tid  = threadIdx.x;
    const int m0   = blockIdx.x * 128;
    const int n0   = blockIdx.y * 128;

    // Each thread stages 32 contiguous bytes (16 bf16) of one row per tile.
    const int row  = tid >> 1;            // 0..127
    const int koff = (tid & 1) * 16;      // 0 or 16 (elements)

    const __bf16* avis = nullptr;
    const __bf16* alm  = nullptr;
    const __bf16* aptr = nullptr;
    if constexpr (GATHER) {
        int t = m0 + row;
        avis = vis + (size_t)bidx[t] * VIS_FLAT + koff;
        alm  = lm  + (size_t)t * LM_DIM + koff;
    } else {
        aptr = A + (size_t)(m0 + row) * K + koff;
    }
    const __bf16* bptr = Wt + (size_t)(n0 + row) * K + koff;

    // Wave/lane geometry for WMMA fragments
    const int lane = tid & 31;
    const int lr   = lane & 15;
    const int lh   = lane >> 4;           // K-half select
    const int wave = tid >> 5;
    const int wm   = (wave >> 1) * 32;    // 0,32,64,96
    const int wn   = (wave & 1) * 64;     // 0,64

    v8f acc[2][4];
    #pragma unroll
    for (int i = 0; i < 2; ++i)
        #pragma unroll
        for (int j = 0; j < 4; ++j)
            acc[i][j] = v8f{};

    auto issue_tile = [&](int buf, int k0) {
        const __bf16* pa;
        if constexpr (GATHER) {
            pa = (k0 < VIS_FLAT) ? (avis + k0) : (alm + (k0 - VIS_FLAT));
        } else {
            pa = aptr + k0;
        }
        const __bf16* pb = bptr + k0;
        __bf16* la = &lds[buf][0][row * 32 + koff];
        __bf16* lb = &lds[buf][1][row * 32 + koff];
        async_b128<0>(pa, la);
        async_b128<16>(pa, la);
        async_b128<0>(pb, lb);
        async_b128<16>(pb, lb);
    };

    const int nsteps = K >> 5;
    issue_tile(0, 0);

    for (int s = 0; s < nsteps; ++s) {
        const int buf = s & 1;
        wait_async0();        // my async writes for `buf` have landed in LDS
        __syncthreads();      // everyone's have; prev compute on buf^1 done

        if (s + 1 < nsteps) issue_tile(buf ^ 1, (s + 1) * 32);  // overlap DMA

        const __bf16* lA = lds[buf][0];
        const __bf16* lB = lds[buf][1];

        // A fragments: lane<16 -> K {0..7,16..23}; lane>=16 -> K {8..15,24..31}
        V16BF afrag[2];
        #pragma unroll
        for (int i = 0; i < 2; ++i) {
            int m = wm + i * 16 + lr;
            afrag[i].h[0] = *(const v8bf*)&lA[m * 32 + lh * 8];
            afrag[i].h[1] = *(const v8bf*)&lA[m * 32 + 16 + lh * 8];
        }
        // B fragments: lane<16 -> K {0..7,16..23} of col n; lane>=16 -> K {8..15,24..31}
        V16BF bfrag[4];
        #pragma unroll
        for (int j = 0; j < 4; ++j) {
            int n = wn + j * 16 + lr;
            bfrag[j].h[0] = *(const v8bf*)&lB[n * 32 + lh * 8];
            bfrag[j].h[1] = *(const v8bf*)&lB[n * 32 + 16 + lh * 8];
        }

        #pragma unroll
        for (int i = 0; i < 2; ++i)
            #pragma unroll
            for (int j = 0; j < 4; ++j)
                acc[i][j] = __builtin_amdgcn_wmma_f32_16x16x32_bf16(
                    false, afrag[i].v, false, bfrag[j].v,
                    (short)0, acc[i][j], false, false);
    }

    // Epilogue: +bias, ReLU, bf16 store.
    // C layout: lanes 0-15: n=lane, VGPR r -> m=r; lanes 16-31: m=8+r.
    #pragma unroll
    for (int j = 0; j < 4; ++j) {
        int n = n0 + wn + j * 16 + lr;
        float bv = bias[n];
        #pragma unroll
        for (int i = 0; i < 2; ++i) {
            #pragma unroll
            for (int r = 0; r < 8; ++r) {
                int m = m0 + wm + i * 16 + lh * 8 + r;
                float v = acc[i][j][r] + bv;
                v = v > 0.0f ? v : 0.0f;
                out[(size_t)m * FF + n] = f2bf(v);
            }
        }
    }
}

// ---------------------------------------------------------------------------
// Kernel 4: final 1024 -> 6 head (tiny; plain f32 FMAs)
// ---------------------------------------------------------------------------
__global__ __launch_bounds__(256)
void head6_kernel(const __bf16* __restrict__ H, const float* __restrict__ W5,
                  const float* __restrict__ b5, float* __restrict__ out) {
    int idx = blockIdx.x * 256 + threadIdx.x;
    if (idx >= TOK * OUTD) return;
    int t = idx / OUTD;
    int j = idx - t * OUTD;
    const __bf16* h = H + (size_t)t * FF;
    float acc = b5[j];
    #pragma unroll 8
    for (int k = 0; k < FF; ++k)
        acc += bf2f(h[k]) * W5[(size_t)k * OUTD + j];
    out[idx] = acc;
}

// ---------------------------------------------------------------------------
// Host side
// ---------------------------------------------------------------------------
extern "C" void kernel_launch(void* const* d_in, const int* in_sizes, int n_in,
                              void* d_out, int out_size, void* d_ws, size_t ws_size,
                              hipStream_t stream) {
    (void)in_sizes; (void)n_in; (void)out_size; (void)ws_size;

    const float* grd  = (const float*)d_in[0];   // [TOK][LM_DIM]
    const float* visf = (const float*)d_in[1];   // [BATCH][N_V][V_DIM]
    const int*   bidx = (const int*)d_in[2];     // [TOK]
    const float* W1   = (const float*)d_in[3];
    const float* b1   = (const float*)d_in[4];
    const float* W2   = (const float*)d_in[5];
    const float* b2   = (const float*)d_in[6];
    const float* W3   = (const float*)d_in[7];
    const float* b3   = (const float*)d_in[8];
    const float* W4   = (const float*)d_in[9];
    const float* b4   = (const float*)d_in[10];
    const float* W5   = (const float*)d_in[11];
    const float* b5   = (const float*)d_in[12];

    char*  ws  = (char*)d_ws;
    size_t off = 0;
    auto carve = [&](size_t bytes) -> void* {
        off = (off + 255) & ~(size_t)255;
        void* p = ws + off;
        off += bytes;
        return p;
    };

    __bf16* Wt1  = (__bf16*)carve((size_t)FF * IN_DIM * 2);   // 75.5 MB -> L2-resident
    __bf16* Wt2  = (__bf16*)carve((size_t)FF * FF * 2);
    __bf16* Wt3  = (__bf16*)carve((size_t)FF * FF * 2);
    __bf16* Wt4  = (__bf16*)carve((size_t)FF * FF * 2);
    __bf16* visb = (__bf16*)carve((size_t)BATCH * VIS_FLAT * 2);
    __bf16* grdb = (__bf16*)carve((size_t)TOK * LM_DIM * 2);
    __bf16* H0   = (__bf16*)carve((size_t)TOK * FF * 2);
    __bf16* H1   = (__bf16*)carve((size_t)TOK * FF * 2);

    // Phase 0: precision conversion (one HBM pass; bf16 W1 then lives in L2)
    {
        int nv = BATCH * VIS_FLAT;
        f32_to_bf16_kernel<<<(nv + 255) / 256, 256, 0, stream>>>(visf, visb, nv);
        int ng = TOK * LM_DIM;
        f32_to_bf16_kernel<<<(ng + 255) / 256, 256, 0, stream>>>(grd, grdb, ng);
        conv_transpose_kernel<<<dim3(IN_DIM / 32, FF / 32), 256, 0, stream>>>(W1, Wt1, IN_DIM, FF);
        conv_transpose_kernel<<<dim3(FF / 32, FF / 32), 256, 0, stream>>>(W2, Wt2, FF, FF);
        conv_transpose_kernel<<<dim3(FF / 32, FF / 32), 256, 0, stream>>>(W3, Wt3, FF, FF);
        conv_transpose_kernel<<<dim3(FF / 32, FF / 32), 256, 0, stream>>>(W4, Wt4, FF, FF);
    }

    // Phase 1: fused gather+concat GEMM (the 155 GFLOP layer)
    dim3 gemm_grid(TOK / 128, FF / 128);   // 16 x 8
    gemm_relu_kernel<true><<<gemm_grid, 256, 0, stream>>>(
        nullptr, visb, grdb, bidx, Wt1, b1, H0, IN_DIM);

    // Phase 2-4: square 1024x1024 layers, ping-pong H buffers
    gemm_relu_kernel<false><<<gemm_grid, 256, 0, stream>>>(
        H0, nullptr, nullptr, nullptr, Wt2, b2, H1, FF);
    gemm_relu_kernel<false><<<gemm_grid, 256, 0, stream>>>(
        H1, nullptr, nullptr, nullptr, Wt3, b3, H0, FF);
    gemm_relu_kernel<false><<<gemm_grid, 256, 0, stream>>>(
        H0, nullptr, nullptr, nullptr, Wt4, b4, H1, FF);

    // Phase 5: tiny 1024->6 head, f32 out
    head6_kernel<<<(TOK * OUTD + 255) / 256, 256, 0, stream>>>(
        H1, W5, b5, (float*)d_out);
}
